// Sacanet_47502338293818
// MI455X (gfx1250) — compile-verified, matrix operand order checked by reading, not run on hardware
//
#include <hip/hip_runtime.h>
#include <hip/hip_bf16.h>
#include <stdint.h>

// ---------------------------------------------------------------------------
// Types / WMMA helpers (gfx1250, wave32, V_WMMA_F32_16X16X32_BF16)
// ---------------------------------------------------------------------------
typedef __bf16 bf16;
typedef bf16  v16bf __attribute__((ext_vector_type(16)));
typedef float v8f   __attribute__((ext_vector_type(8)));
typedef int   v4i   __attribute__((vector_size(16)));

#define WMMA_BF16 __builtin_amdgcn_wmma_f32_16x16x32_bf16

// CDNA5 async global->LDS staging (GLOBAL_LOAD_ASYNC_TO_LDS_B128, ASYNCcnt).
#if defined(__has_builtin)
#if __has_builtin(__builtin_amdgcn_global_load_async_to_lds_b128) && \
    __has_builtin(__builtin_amdgcn_s_wait_asynccnt)
#define USE_ASYNC_LDS 1
#endif
#endif
#ifndef USE_ASYNC_LDS
#define USE_ASYNC_LDS 0
#endif

static __device__ __forceinline__ v8f vzero() {
    union { v8f v; float f[8]; } u;
#pragma unroll
    for (int i = 0; i < 8; ++i) u.f[i] = 0.f;
    return u.v;
}

// copy 16 bf16 (32B) as two 16B vectors through VGPRs
static __device__ __forceinline__ void cp16(bf16* __restrict__ d, const bf16* __restrict__ s) {
    reinterpret_cast<uint4*>(d)[0] = reinterpret_cast<const uint4*>(s)[0];
    reinterpret_cast<uint4*>(d)[1] = reinterpret_cast<const uint4*>(s)[1];
}

// stage 16 bf16 (32B) global -> LDS; async path bypasses VGPRs (2x b128)
static __device__ __forceinline__ void stage32(bf16* __restrict__ d, const bf16* __restrict__ s) {
#if USE_ASYNC_LDS
    __builtin_amdgcn_global_load_async_to_lds_b128(
        (__attribute__((address_space(1))) v4i*)(s),
        (__attribute__((address_space(3))) v4i*)(d), 0, 0);
    __builtin_amdgcn_global_load_async_to_lds_b128(
        (__attribute__((address_space(1))) v4i*)(s),
        (__attribute__((address_space(3))) v4i*)(d), 16, 0);
#else
    cp16(d, s);
#endif
}

static __device__ __forceinline__ void stage_wait() {
#if USE_ASYNC_LDS
    __builtin_amdgcn_s_wait_asynccnt(0);
#endif
}

// Load a 16x32 bf16 A/B fragment from a row-major [row][K] tile.
// ISA 7.12.2: lanes 0-15 hold K 0..7 & 16..23, lanes 16-31 hold K 8..15 & 24..31.
static __device__ __forceinline__ v16bf ldfrag(const bf16* base, int tilerow, int stride, int k0) {
    int lane = threadIdx.x & 31;
    const bf16* p = base + (size_t)(tilerow + (lane & 15)) * stride + k0 + ((lane >> 4) << 3);
    union { v16bf v; uint4 q[2]; } u;
    u.q[0] = *reinterpret_cast<const uint4*>(p);
    u.q[1] = *reinterpret_cast<const uint4*>(p + 16);
    return u.v;
}

// patch p (0..255) = b*64 + py*8 + px ; token l (0..255) = i*16 + j -> global pixel
static __device__ __forceinline__ int patch_pix(int p, int l) {
    int b = p >> 6, py = (p >> 3) & 7, px = p & 7;
    int i = l >> 4, j = l & 15;
    return ((b * 128 + py * 16 + i) * 128) + px * 16 + j;
}

// ---------------------------------------------------------------------------
// Prep kernels (BN folding, weight transpose + bf16 conversion, RPE gather)
// ---------------------------------------------------------------------------
__global__ void k_bnprep(const float* g, const float* be, const float* m, const float* v,
                         float* sc, float* sh, int C) {
    int t = threadIdx.x + blockIdx.x * blockDim.x;
    if (t < C) {
        float inv = g[t] * rsqrtf(v[t] + 1e-5f);
        sc[t] = inv;
        sh[t] = be[t] - m[t] * inv;
    }
}

__global__ void k_f2b(const float* s, bf16* d, int n) {
    int t = threadIdx.x + blockIdx.x * blockDim.x;
    if (t < n) d[t] = (bf16)s[t];
}

// bottleneck_w (128,480,3,3) -> Wt[o][tap*480+ci]
__global__ void k_wtr_bott(const float* s, bf16* d) {
    int idx = threadIdx.x + blockIdx.x * blockDim.x;
    if (idx >= 128 * 4320) return;
    int o = idx / 4320, k = idx % 4320;
    int tap = k / 480, ci = k % 480;
    d[idx] = (bf16)s[(o * 480 + ci) * 9 + tap];
}

// cat_w (128,256,3,3) -> Wt[o][tap*256+c]
__global__ void k_wtr_cat(const float* s, bf16* d) {
    int idx = threadIdx.x + blockIdx.x * blockDim.x;
    if (idx >= 128 * 2304) return;
    int o = idx / 2304, k = idx % 2304;
    int tap = k / 256, c = k % 256;
    d[idx] = (bf16)s[(o * 256 + c) * 9 + tap];
}

// dec2_w (6,64) -> padded bf16 [16][64]
__global__ void k_wdec2(const float* s, bf16* d) {
    int idx = threadIdx.x + blockIdx.x * blockDim.x;
    if (idx >= 16 * 64) return;
    int o = idx / 64, i = idx % 64;
    d[idx] = (bf16)(o < 6 ? s[o * 64 + i] : 0.f);
}

// RPEb[l][m][d] = rpe_table[d][bucket(l,m)] ; bucket from 16x16 relative coords
__global__ void k_rpeb(const float* rpe, bf16* d) {
    int l = blockIdx.x, m = threadIdx.x;
    int i0 = l >> 4, j0 = l & 15, i1 = m >> 4, j1 = m & 15;
    int bucket = (i0 - i1 + 15) * 31 + (j0 - j1 + 15);
    size_t o = ((size_t)(l * 256 + m)) * 64;
    for (int dd = 0; dd < 64; ++dd) d[o + dd] = (bf16)rpe[dd * 961 + bucket];
}

__global__ void k_zero(float* p, int n) {
    for (int i = threadIdx.x; i < n; i += 256) p[i] = 0.f;
}

// ---------------------------------------------------------------------------
// Bottleneck conv3x3 (480->128) implicit GEMM, fused BN+ReLU, bf16 out into
// comb[pix][128..255] (feats slice). One block per image row (M=128 pixels).
// ---------------------------------------------------------------------------
__global__ __launch_bounds__(256, 1) void k_conv3_480(
    const float* __restrict__ x, const bf16* __restrict__ Wt,
    const float* __restrict__ scsh, bf16* __restrict__ comb) {
    __shared__ bf16 As[128 * 32];
    __shared__ bf16 Bs[128 * 32];
    int blk = blockIdx.x;
    int b = blk >> 7, y = blk & 127;
    int t = threadIdx.x, wid = t >> 5, lane = t & 31;
    v8f acc[8];
#pragma unroll
    for (int i = 0; i < 8; ++i) acc[i] = vzero();
    int xA = t & 127, kh2 = t >> 7;  // pixel, K half (0/1)
    for (int kc = 0; kc < 135; ++kc) {
        int tap = kc / 15, ci0 = (kc % 15) * 32;
        int dy = tap / 3 - 1, dx = tap % 3 - 1;
        int yy = y + dy, xx = xA + dx;
        bool ok = (yy >= 0) && (yy < 128) && (xx >= 0) && (xx < 128);
        const float* src = x + ((size_t)(b * 480 + ci0 + kh2 * 16) * 128 + (ok ? yy : 0)) * 128 + (ok ? xx : 0);
#pragma unroll
        for (int j = 0; j < 16; ++j) {
            float v = ok ? src[(size_t)j * 16384] : 0.f;
            As[xA * 32 + kh2 * 16 + j] = (bf16)v;
        }
        {
            int n = t >> 1, kh = t & 1;
            stage32(&Bs[n * 32 + kh * 16], Wt + (size_t)n * 4320 + kc * 32 + kh * 16);
            if (kc + 1 < 135)
                __builtin_prefetch(Wt + (size_t)n * 4320 + (kc + 1) * 32 + kh * 16, 0, 0);
        }
        stage_wait();
        __syncthreads();
        v16bf a = ldfrag(As, wid * 16, 32, 0);
#pragma unroll
        for (int nt = 0; nt < 8; ++nt) {
            v16bf bb = ldfrag(Bs, nt * 16, 32, 0);
            acc[nt] = WMMA_BF16(false, a, false, bb, (short)0, acc[nt], false, false);
        }
        __syncthreads();
    }
    int mo = (lane >> 4) * 8;
#pragma unroll
    for (int nt = 0; nt < 8; ++nt) {
        int o = nt * 16 + (lane & 15);
        float s = scsh[o], h0 = scsh[128 + o];
        const float* af = (const float*)&acc[nt];
#pragma unroll
        for (int r = 0; r < 8; ++r) {
            int xO = wid * 16 + r + mo;
            float v = fmaxf(af[r] * s + h0, 0.f);
            size_t pix = (size_t)b * 16384 + y * 128 + xO;
            comb[pix * 256 + 128 + o] = (bf16)v;
        }
    }
}

// ---------------------------------------------------------------------------
// cat conv3x3 (256->128) implicit GEMM over comb, fused BN+ReLU, fp32 NCHW out
// ---------------------------------------------------------------------------
__global__ __launch_bounds__(256, 1) void k_conv3_cat(
    const bf16* __restrict__ comb, const bf16* __restrict__ Wt,
    const float* __restrict__ scsh, float* __restrict__ out) {
    __shared__ bf16 As[128 * 32];
    __shared__ bf16 Bs[128 * 32];
    int blk = blockIdx.x;
    int b = blk >> 7, y = blk & 127;
    int t = threadIdx.x, wid = t >> 5, lane = t & 31;
    v8f acc[8];
#pragma unroll
    for (int i = 0; i < 8; ++i) acc[i] = vzero();
    int xA = t >> 1, hx = t & 1;
    for (int kc = 0; kc < 72; ++kc) {
        int tap = kc >> 3, c0 = (kc & 7) * 32;
        int dy = tap / 3 - 1, dx = tap % 3 - 1;
        int yy = y + dy, xx = xA + dx;
        bool ok = (yy >= 0) && (yy < 128) && (xx >= 0) && (xx < 128);
        if (ok) {
            size_t pix = (size_t)b * 16384 + yy * 128 + xx;
            stage32(&As[xA * 32 + hx * 16], comb + pix * 256 + c0 + hx * 16);
        } else {
#pragma unroll
            for (int j = 0; j < 16; ++j) As[xA * 32 + hx * 16 + j] = (bf16)0.f;
        }
        {
            int n = t >> 1, kh = t & 1;
            stage32(&Bs[n * 32 + kh * 16], Wt + (size_t)n * 2304 + kc * 32 + kh * 16);
            if (kc + 1 < 72)
                __builtin_prefetch(Wt + (size_t)n * 2304 + (kc + 1) * 32 + kh * 16, 0, 0);
        }
        stage_wait();
        __syncthreads();
        v16bf a = ldfrag(As, wid * 16, 32, 0);
#pragma unroll
        for (int nt = 0; nt < 8; ++nt) {
            v16bf bb = ldfrag(Bs, nt * 16, 32, 0);
            acc[nt] = WMMA_BF16(false, a, false, bb, (short)0, acc[nt], false, false);
        }
        __syncthreads();
    }
    int mo = (lane >> 4) * 8;
#pragma unroll
    for (int nt = 0; nt < 8; ++nt) {
        int o = nt * 16 + (lane & 15);
        float s = scsh[o], h0 = scsh[128 + o];
        const float* af = (const float*)&acc[nt];
#pragma unroll
        for (int r = 0; r < 8; ++r) {
            int xO = wid * 16 + r + mo;
            float v = fmaxf(af[r] * s + h0, 0.f);
            out[(((size_t)b * 128 + o) * 128 + y) * 128 + xO] = v;
        }
    }
}

// ---------------------------------------------------------------------------
// Generic row-major bf16 GEMM: out[M x (NT16*16)] = A[M x Ka] * Bt[N][Ka]^T
// mode 0: bf16 out with fused BN+ReLU.  mode 1: dec2 -> fp32 NCHW preds1+bias.
// ---------------------------------------------------------------------------
template <int NT16>
__global__ __launch_bounds__(256, 1) void k_gemm_rm(
    const bf16* __restrict__ A, int lda, int aoff,
    const bf16* __restrict__ Bt, int Ka,
    const float* __restrict__ sc, const float* __restrict__ sh,
    int mode, bf16* __restrict__ outb, int ldo, int ooff,
    float* __restrict__ outf, const float* __restrict__ bias) {
    __shared__ bf16 As[128 * 32];
    __shared__ bf16 Bs[NT16 * 16 * 32];
    int t = threadIdx.x, wid = t >> 5, lane = t & 31;
    size_t row0 = (size_t)blockIdx.x * 128;
    v8f acc[NT16];
#pragma unroll
    for (int i = 0; i < NT16; ++i) acc[i] = vzero();
    int nkc = Ka >> 5;
    for (int kc = 0; kc < nkc; ++kc) {
        {
            int r = t >> 1, kh = t & 1;
            stage32(&As[r * 32 + kh * 16], A + (row0 + r) * lda + aoff + kc * 32 + kh * 16);
        }
        for (int cid = t; cid < NT16 * 32; cid += 256) {
            int n = cid >> 1, kh = cid & 1;
            stage32(&Bs[n * 32 + kh * 16], Bt + (size_t)n * Ka + kc * 32 + kh * 16);
        }
        stage_wait();
        __syncthreads();
        v16bf a = ldfrag(As, wid * 16, 32, 0);
#pragma unroll
        for (int nt = 0; nt < NT16; ++nt) {
            v16bf bb = ldfrag(Bs, nt * 16, 32, 0);
            acc[nt] = WMMA_BF16(false, a, false, bb, (short)0, acc[nt], false, false);
        }
        __syncthreads();
    }
    int mo = (lane >> 4) * 8;
    if (mode == 0) {
#pragma unroll
        for (int nt = 0; nt < NT16; ++nt) {
            int n = nt * 16 + (lane & 15);
            float s = sc[n], h0 = sh[n];
            const float* af = (const float*)&acc[nt];
#pragma unroll
            for (int r = 0; r < 8; ++r) {
                size_t row = row0 + wid * 16 + r + mo;
                outb[row * ldo + ooff + n] = (bf16)fmaxf(af[r] * s + h0, 0.f);
            }
        }
    } else {
        int n = lane & 15;
        const float* af = (const float*)&acc[0];
        if (n < 6) {
            float bsv = bias[n];
#pragma unroll
            for (int r = 0; r < 8; ++r) {
                size_t row = row0 + wid * 16 + r + mo;
                int b = (int)(row >> 14), l = (int)(row & 16383);
                outf[((size_t)(b * 6 + n)) * 16384 + l] = af[r] + bsv;
            }
        }
    }
}

// ---------------------------------------------------------------------------
// Full-resolution spatial-gather: per (b,class) plane softmax stats
// ---------------------------------------------------------------------------
__global__ __launch_bounds__(256) void k_sm_stats(const float* __restrict__ preds,
                                                  float* gmax, float* gsum) {
    __shared__ float red[256];
    int plane = blockIdx.x;
    const float* p = preds + (size_t)plane * 16384;
    int t = threadIdx.x;
    float m = -1e30f;
    for (int i = t; i < 16384; i += 256) m = fmaxf(m, p[i]);
    red[t] = m;
    __syncthreads();
    for (int s = 128; s > 0; s >>= 1) { if (t < s) red[t] = fmaxf(red[t], red[t + s]); __syncthreads(); }
    float mx = red[0];
    __syncthreads();
    float su = 0.f;
    for (int i = t; i < 16384; i += 256) su += __expf(p[i] - mx);
    red[t] = su;
    __syncthreads();
    for (int s = 128; s > 0; s >>= 1) { if (t < s) red[t] += red[t + s]; __syncthreads(); }
    if (t == 0) { gmax[plane] = mx; gsum[plane] = red[0]; }
}

// ctxg[b][n][c] += sum_l softmax(p)[l] * feats[l][c]  (partial, atomics)
__global__ __launch_bounds__(256) void k_gather_global(
    const float* __restrict__ preds, const float* __restrict__ gmax,
    const float* __restrict__ gsum, const bf16* __restrict__ comb,
    float* __restrict__ ctxg) {
    int blk = blockIdx.x;       // 64 = 4 b * 16 segments
    int b = blk >> 4, seg = blk & 15;
    int t = threadIdx.x, c = t & 127, hb = t >> 7;
    int l0 = seg * 1024 + hb * 512;
    float mv[6], sv[6];
#pragma unroll
    for (int n = 0; n < 6; ++n) { mv[n] = gmax[b * 6 + n]; sv[n] = gsum[b * 6 + n]; }
    float acc[6] = {0.f, 0.f, 0.f, 0.f, 0.f, 0.f};
    for (int l = l0; l < l0 + 512; ++l) {
        float f = (float)comb[((size_t)(b * 16384 + l)) * 256 + 128 + c];
#pragma unroll
        for (int n = 0; n < 6; ++n)
            acc[n] += __expf(preds[(size_t)(b * 6 + n) * 16384 + l] - mv[n]) * f;
    }
#pragma unroll
    for (int n = 0; n < 6; ++n)
        atomicAdd(&ctxg[(b * 6 + n) * 128 + c], acc[n] / sv[n]);
}

// per-pixel argmax class -> recovered context ctxpix[pix][128] (bf16)
__global__ __launch_bounds__(256) void k_recover_global(
    const float* __restrict__ preds, const float* __restrict__ ctxg,
    bf16* __restrict__ ctxpix) {
    int t = threadIdx.x;
    int pix = blockIdx.x * 64 + (t >> 2);
    int q4 = t & 3;
    int b = pix >> 14, l = pix & 16383;
    float best = -1e30f; int bi = 0;
#pragma unroll
    for (int n = 0; n < 6; ++n) {
        float v = preds[(size_t)(b * 6 + n) * 16384 + l];
        if (v > best) { best = v; bi = n; }
    }
    const float* src = ctxg + (b * 6 + bi) * 128 + q4 * 32;
    bf16* dst = ctxpix + (size_t)pix * 128 + q4 * 32;
#pragma unroll
    for (int j = 0; j < 32; ++j) dst[j] = (bf16)src[j];
}

// ---------------------------------------------------------------------------
// Patch-level gather+recover -> plc (key feats), one block per patch
// ---------------------------------------------------------------------------
__global__ __launch_bounds__(256) void k_patch_gather(
    const float* __restrict__ preds, const bf16* __restrict__ comb,
    bf16* __restrict__ plc) {
    __shared__ float lp[6 * 256];
    __shared__ float prob[6 * 256];
    __shared__ float red[256];
    __shared__ int idx[256];
    __shared__ float sp[6 * 256];
    __shared__ float ctxp[6 * 128];
    int p = blockIdx.x, t = threadIdx.x;
    int b = p >> 6;
    int pix = patch_pix(p, t);
    int l16 = pix & 16383;
    float best = -1e30f; int bi = 0;
#pragma unroll
    for (int n = 0; n < 6; ++n) {
        float v = preds[(size_t)(b * 6 + n) * 16384 + l16];
        lp[n * 256 + t] = v;
        if (v > best) { best = v; bi = n; }
    }
    idx[t] = bi;
    __syncthreads();
    for (int n = 0; n < 6; ++n) {
        red[t] = lp[n * 256 + t];
        __syncthreads();
        for (int s = 128; s; s >>= 1) { if (t < s) red[t] = fmaxf(red[t], red[t + s]); __syncthreads(); }
        float mx = red[0];
        __syncthreads();
        float e = __expf(lp[n * 256 + t] - mx);
        red[t] = e;
        __syncthreads();
        for (int s = 128; s; s >>= 1) { if (t < s) red[t] += red[t + s]; __syncthreads(); }
        prob[n * 256 + t] = e / red[0];
        __syncthreads();
    }
    int c = t & 127, hb = t >> 7;
    float a0 = 0, a1 = 0, a2 = 0, a3 = 0, a4 = 0, a5 = 0;
    for (int l = hb * 128; l < hb * 128 + 128; ++l) {
        float f = (float)comb[(size_t)patch_pix(p, l) * 256 + 128 + c];
        a0 += prob[0 * 256 + l] * f; a1 += prob[1 * 256 + l] * f;
        a2 += prob[2 * 256 + l] * f; a3 += prob[3 * 256 + l] * f;
        a4 += prob[4 * 256 + l] * f; a5 += prob[5 * 256 + l] * f;
    }
    sp[0 * 256 + hb * 128 + c] = a0; sp[1 * 256 + hb * 128 + c] = a1;
    sp[2 * 256 + hb * 128 + c] = a2; sp[3 * 256 + hb * 128 + c] = a3;
    sp[4 * 256 + hb * 128 + c] = a4; sp[5 * 256 + hb * 128 + c] = a5;
    __syncthreads();
    if (hb == 0) {
#pragma unroll
        for (int n = 0; n < 6; ++n) ctxp[n * 128 + c] = sp[n * 256 + c] + sp[n * 256 + 128 + c];
    }
    __syncthreads();
    int mi = idx[t];
    size_t o = (size_t)pix * 128;
    for (int cc = 0; cc < 128; ++cc) plc[o + cc] = (bf16)ctxp[mi * 128 + cc];
}

// ---------------------------------------------------------------------------
// Channel-attention gate -> qf[pix][128] (bf16), one block per patch
// ---------------------------------------------------------------------------
__global__ __launch_bounds__(256) void k_cattn(
    const bf16* __restrict__ comb, const float* __restrict__ w1,
    const float* __restrict__ w2, bf16* __restrict__ qf) {
    __shared__ float s1[256], s2[256], favg[128], fmx[128], h8a[8], h8m[8], gate[128];
    int p = blockIdx.x, t = threadIdx.x;
    int c = t & 127, hb = t >> 7;
    float sum = 0.f, mx = -1e30f;
    for (int l = hb * 128; l < hb * 128 + 128; ++l) {
        float f = (float)comb[(size_t)patch_pix(p, l) * 256 + 128 + c];
        sum += f; mx = fmaxf(mx, f);
    }
    s1[t] = sum; s2[t] = mx;
    __syncthreads();
    if (hb == 0) {
        favg[c] = (s1[c] + s1[c + 128]) * (1.f / 256.f);
        fmx[c] = fmaxf(s2[c], s2[c + 128]);
    }
    __syncthreads();
    if (t < 8) {
        float aa = 0, am = 0;
        for (int i = 0; i < 128; ++i) { float w = w1[t * 128 + i]; aa += w * favg[i]; am += w * fmx[i]; }
        h8a[t] = fmaxf(aa, 0.f); h8m[t] = fmaxf(am, 0.f);
    }
    __syncthreads();
    if (t < 128) {
        float ga = 0, gm = 0;
#pragma unroll
        for (int j = 0; j < 8; ++j) { float w = w2[t * 8 + j]; ga += w * h8a[j]; gm += w * h8m[j]; }
        gate[t] = 1.f / (1.f + __expf(-(ga + gm)));
    }
    __syncthreads();
    for (int l = hb * 128; l < hb * 128 + 128; ++l) {
        size_t o = (size_t)patch_pix(p, l);
        qf[o * 128 + c] = (bf16)((float)comb[o * 256 + 128 + c] * gate[c]);
    }
}

// ---------------------------------------------------------------------------
// RPE bias GEMM: POS[l][patch][m] = q[patch@l] . G[l][m][:]   (bf16 WMMA)
// grid (l=256, mblk=2, nblk=2), 128 patches x 128 m per block, K=64
// ---------------------------------------------------------------------------
__global__ __launch_bounds__(256, 1) void k_pos_gemm(
    const bf16* __restrict__ q, const bf16* __restrict__ rpeb, bf16* __restrict__ POS) {
    __shared__ bf16 As[128 * 32];
    __shared__ bf16 Bs[128 * 32];
    int l = blockIdx.x, mblk = blockIdx.y, nblk = blockIdx.z;
    int t = threadIdx.x, wid = t >> 5, lane = t & 31;
    v8f acc[8];
#pragma unroll
    for (int i = 0; i < 8; ++i) acc[i] = vzero();
    for (int kc = 0; kc < 2; ++kc) {
        {
            int row = t >> 1, kh = t & 1;
            int patch = mblk * 128 + row;
            stage32(&As[row * 32 + kh * 16],
                    q + (size_t)patch_pix(patch, l) * 64 + kc * 32 + kh * 16);
        }
        {
            int n = t >> 1, kh = t & 1;
            stage32(&Bs[n * 32 + kh * 16],
                    rpeb + ((size_t)(l * 256 + nblk * 128 + n)) * 64 + kc * 32 + kh * 16);
        }
        stage_wait();
        __syncthreads();
        v16bf a = ldfrag(As, wid * 16, 32, 0);
#pragma unroll
        for (int nt = 0; nt < 8; ++nt) {
            v16bf bb = ldfrag(Bs, nt * 16, 32, 0);
            acc[nt] = WMMA_BF16(false, a, false, bb, (short)0, acc[nt], false, false);
        }
        __syncthreads();
    }
    int mo = (lane >> 4) * 8;
#pragma unroll
    for (int nt = 0; nt < 8; ++nt) {
        int m = nblk * 128 + nt * 16 + (lane & 15);
        const float* af = (const float*)&acc[nt];
#pragma unroll
        for (int r = 0; r < 8; ++r) {
            int patch = mblk * 128 + wid * 16 + r + mo;
            POS[((size_t)(l * 256 + patch)) * 256 + m] = (bf16)af[r];
        }
    }
}

// ---------------------------------------------------------------------------
// Attention: one block per (patch, row-half). Q/K/V^T + full score rows in LDS
// (272KB dynamic LDS -- gfx1250 WGP has 320KB). WMMA for QK^T and attn*V.
// ---------------------------------------------------------------------------
__global__ __launch_bounds__(256, 1) void k_attn(
    const bf16* __restrict__ q, const bf16* __restrict__ kk,
    const bf16* __restrict__ vv, const bf16* __restrict__ POS,
    bf16* __restrict__ ctxa) {
    extern __shared__ char smem[];
    bf16*  Qs = (bf16*)smem;                        // 128*64  (16 KB)
    bf16*  Ks = (bf16*)(smem + 16384);              // 256*64  (32 KB)
    bf16*  Vt = (bf16*)(smem + 16384 + 32768);      // 64*256  (32 KB)
    float* Sf = (float*)(smem + 81920);             // 128*256 (128 KB)
    bf16*  Sb = (bf16*)(smem + 81920 + 131072);     // 128*256 (64 KB)
    int p = blockIdx.x >> 1, hb = blockIdx.x & 1;
    int t = threadIdx.x, wid = t >> 5, lane = t & 31;

    for (int cid = t; cid < 512; cid += 256) {
        int r = cid >> 2, c4 = cid & 3;
        int pix = patch_pix(p, hb * 128 + r);
        stage32(&Qs[r * 64 + c4 * 16], q + (size_t)pix * 64 + c4 * 16);
    }
    for (int cid = t; cid < 1024; cid += 256) {
        int m = cid >> 2, c4 = cid & 3;
        int pix = patch_pix(p, m);
        stage32(&Ks[m * 64 + c4 * 16], kk + (size_t)pix * 64 + c4 * 16);
    }
    for (int ix = t; ix < 16384; ix += 256) {
        int m = ix >> 6, d = ix & 63;
        Vt[d * 256 + m] = vv[(size_t)patch_pix(p, m) * 64 + d];
    }
    stage_wait();
    __syncthreads();

    v8f acc[16];
#pragma unroll
    for (int i = 0; i < 16; ++i) acc[i] = vzero();
#pragma unroll
    for (int kc = 0; kc < 2; ++kc) {
        v16bf a = ldfrag(Qs, wid * 16, 64, kc * 32);
#pragma unroll
        for (int nt = 0; nt < 16; ++nt) {
            v16bf bb = ldfrag(Ks, nt * 16, 64, kc * 32);
            acc[nt] = WMMA_BF16(false, a, false, bb, (short)0, acc[nt], false, false);
        }
    }
    const float scale = 0.125f;  // AC^-0.5
    int mo = (lane >> 4) * 8;
#pragma unroll
    for (int nt = 0; nt < 16; ++nt) {
        int m = nt * 16 + (lane & 15);
        const float* af = (const float*)&acc[nt];
#pragma unroll
        for (int r = 0; r < 8; ++r) {
            int lloc = wid * 16 + r + mo;
            int l = hb * 128 + lloc;
            float pos = (float)POS[((size_t)(l * 256 + p)) * 256 + m];
            Sf[lloc * 256 + m] = (af[r] + pos) * scale;
        }
    }
    __syncthreads();
    if (t < 128) {
        float mx = -1e30f;
        for (int m = 0; m < 256; ++m) mx = fmaxf(mx, Sf[t * 256 + m]);
        float su = 0.f;
        for (int m = 0; m < 256; ++m) { float e = __expf(Sf[t * 256 + m] - mx); Sf[t * 256 + m] = e; su += e; }
        float inv = 1.f / su;
        for (int m = 0; m < 256; ++m) Sb[t * 256 + m] = (bf16)(Sf[t * 256 + m] * inv);
    }
    __syncthreads();

    v8f c2[4];
#pragma unroll
    for (int i = 0; i < 4; ++i) c2[i] = vzero();
#pragma unroll
    for (int kc = 0; kc < 8; ++kc) {
        v16bf a = ldfrag(Sb, wid * 16, 256, kc * 32);
#pragma unroll
        for (int nt = 0; nt < 4; ++nt) {
            v16bf bb = ldfrag(Vt, nt * 16, 256, kc * 32);
            c2[nt] = WMMA_BF16(false, a, false, bb, (short)0, c2[nt], false, false);
        }
    }
#pragma unroll
    for (int nt = 0; nt < 4; ++nt) {
        int d = nt * 16 + (lane & 15);
        const float* af = (const float*)&c2[nt];
#pragma unroll
        for (int r = 0; r < 8; ++r) {
            int l = hb * 128 + wid * 16 + r + mo;
            ctxa[(size_t)patch_pix(p, l) * 64 + d] = (bf16)af[r];
        }
    }
}

// ---------------------------------------------------------------------------
// Workspace layout (bytes, all 256-aligned)
// ---------------------------------------------------------------------------
static constexpr size_t OFF_COMB   = 0;                                       // 65536*256 bf16
static constexpr size_t OFF_WTB    = OFF_COMB   + (size_t)65536 * 256 * 2;    // 128*4320 bf16
static constexpr size_t OFF_WTC    = OFF_WTB    + (size_t)128 * 4320 * 2;     // 128*2304 bf16
static constexpr size_t OFF_WD1    = OFF_WTC    + (size_t)128 * 2304 * 2;     // 64*128
static constexpr size_t OFF_WD2    = OFF_WD1    + 8192 * 2;                   // 16*64
static constexpr size_t OFF_WQ1    = OFF_WD2    + 1024 * 2;
static constexpr size_t OFF_WQ2    = OFF_WQ1    + 8192 * 2;
static constexpr size_t OFF_WK1    = OFF_WQ2    + 4096 * 2;
static constexpr size_t OFF_WK2    = OFF_WK1    + 8192 * 2;
static constexpr size_t OFF_WV     = OFF_WK2    + 4096 * 2;
static constexpr size_t OFF_WO     = OFF_WV     + 8192 * 2;
static constexpr size_t OFF_SC     = OFF_WO     + 8192 * 2;                   // 2048 f32 (BN sc/sh)
static constexpr size_t OFF_H      = OFF_SC     + 2048 * 4;                   // 65536*64 bf16
static constexpr size_t OFF_PLC    = OFF_H      + (size_t)65536 * 64 * 2;
static constexpr size_t OFF_QF     = OFF_PLC    + (size_t)65536 * 128 * 2;
static constexpr size_t OFF_CTXPIX = OFF_QF     + (size_t)65536 * 128 * 2;
static constexpr size_t OFF_T64    = OFF_CTXPIX + (size_t)65536 * 128 * 2;
static constexpr size_t OFF_QB     = OFF_T64    + (size_t)65536 * 64 * 2;
static constexpr size_t OFF_KB     = OFF_QB     + (size_t)65536 * 64 * 2;
static constexpr size_t OFF_VB     = OFF_KB     + (size_t)65536 * 64 * 2;
static constexpr size_t OFF_CTXA   = OFF_VB     + (size_t)65536 * 64 * 2;
static constexpr size_t OFF_RPEB   = OFF_CTXA   + (size_t)65536 * 64 * 2;     // 256*256*64 bf16
static constexpr size_t OFF_POS    = OFF_RPEB   + (size_t)256 * 256 * 64 * 2; // 256*256*256 bf16
static constexpr size_t OFF_CTXG   = OFF_POS    + (size_t)256 * 256 * 256 * 2;// 3072 f32
static constexpr size_t OFF_GST    = OFF_CTXG   + 3072 * 4;                   // 48 f32

extern "C" void kernel_launch(void* const* d_in, const int* in_sizes, int n_in,
                              void* d_out, int out_size, void* d_ws, size_t ws_size,
                              hipStream_t stream) {
    (void)in_sizes; (void)n_in; (void)out_size; (void)ws_size;
    // Flattened input order (setup_inputs dict insertion order, depth-first):
    // 0:x 1:bott_w 2-5:bott_bn 6:dec1_w 7-10:dec1_bn 11:dec2_w 12:dec2_b
    // 13:ca_w1 14:ca_w2 15:q1_w 16-19:q1_bn 20:q2_w 21-24:q2_bn 25:k1_w
    // 26-29:k1_bn 30:k2_w 31-34:k2_bn 35:v_w 36-39:v_bn 40:out_w 41-44:out_bn
    // 45:rpe_table 46:cat_w 47-50:cat_bn
    auto F = [&](int i) { return (const float*)d_in[i]; };

    char* ws = (char*)d_ws;
    bf16*  COMB   = (bf16*)(ws + OFF_COMB);
    bf16*  WTB    = (bf16*)(ws + OFF_WTB);
    bf16*  WTC    = (bf16*)(ws + OFF_WTC);
    bf16*  WD1    = (bf16*)(ws + OFF_WD1);
    bf16*  WD2    = (bf16*)(ws + OFF_WD2);
    bf16*  WQ1    = (bf16*)(ws + OFF_WQ1);
    bf16*  WQ2    = (bf16*)(ws + OFF_WQ2);
    bf16*  WK1    = (bf16*)(ws + OFF_WK1);
    bf16*  WK2    = (bf16*)(ws + OFF_WK2);
    bf16*  WV     = (bf16*)(ws + OFF_WV);
    bf16*  WO     = (bf16*)(ws + OFF_WO);
    float* SC     = (float*)(ws + OFF_SC);
    bf16*  HBUF   = (bf16*)(ws + OFF_H);
    bf16*  PLC    = (bf16*)(ws + OFF_PLC);
    bf16*  QF     = (bf16*)(ws + OFF_QF);
    bf16*  CTXPIX = (bf16*)(ws + OFF_CTXPIX);
    bf16*  T64    = (bf16*)(ws + OFF_T64);
    bf16*  QB     = (bf16*)(ws + OFF_QB);
    bf16*  KB     = (bf16*)(ws + OFF_KB);
    bf16*  VB     = (bf16*)(ws + OFF_VB);
    bf16*  CTXA   = (bf16*)(ws + OFF_CTXA);
    bf16*  RPEB   = (bf16*)(ws + OFF_RPEB);
    bf16*  POSB   = (bf16*)(ws + OFF_POS);
    float* CTXG   = (float*)(ws + OFF_CTXG);
    float* GMAX   = (float*)(ws + OFF_GST);
    float* GSUM   = GMAX + 24;

    float* OUT2 = (float*)d_out;                                   // preds2 NCHW
    float* P1   = (float*)d_out + (size_t)4 * 128 * 128 * 128;     // preds1 NCHW

    // ---- prep: BN folding (sc at offsets, sh = sc+C slot) ----
    k_bnprep<<<1, 128, 0, stream>>>(F(2),  F(3),  F(4),  F(5),  SC + 0,    SC + 128,  128); // bott
    k_bnprep<<<1, 128, 0, stream>>>(F(7),  F(8),  F(9),  F(10), SC + 256,  SC + 320,  64);  // dec1
    k_bnprep<<<1, 128, 0, stream>>>(F(16), F(17), F(18), F(19), SC + 384,  SC + 448,  64);  // q1
    k_bnprep<<<1, 128, 0, stream>>>(F(21), F(22), F(23), F(24), SC + 512,  SC + 576,  64);  // q2
    k_bnprep<<<1, 128, 0, stream>>>(F(26), F(27), F(28), F(29), SC + 640,  SC + 704,  64);  // k1
    k_bnprep<<<1, 128, 0, stream>>>(F(31), F(32), F(33), F(34), SC + 768,  SC + 832,  64);  // k2
    k_bnprep<<<1, 128, 0, stream>>>(F(36), F(37), F(38), F(39), SC + 896,  SC + 960,  64);  // v
    k_bnprep<<<1, 128, 0, stream>>>(F(41), F(42), F(43), F(44), SC + 1024, SC + 1152, 128); // out
    k_bnprep<<<1, 128, 0, stream>>>(F(47), F(48), F(49), F(50), SC + 1280, SC + 1408, 128); // cat

    // ---- prep: weights -> bf16 (transposed for conv taps) ----
    k_wtr_bott<<<(128 * 4320 + 255) / 256, 256, 0, stream>>>(F(1), WTB);
    k_wtr_cat<<<(128 * 2304 + 255) / 256, 256, 0, stream>>>(F(46), WTC);
    k_f2b<<<(8192 + 255) / 256, 256, 0, stream>>>(F(6),  WD1, 8192);
    k_wdec2<<<4, 256, 0, stream>>>(F(11), WD2);
    k_f2b<<<(8192 + 255) / 256, 256, 0, stream>>>(F(15), WQ1, 8192);
    k_f2b<<<(4096 + 255) / 256, 256, 0, stream>>>(F(20), WQ2, 4096);
    k_f2b<<<(8192 + 255) / 256, 256, 0, stream>>>(F(25), WK1, 8192);
    k_f2b<<<(4096 + 255) / 256, 256, 0, stream>>>(F(30), WK2, 4096);
    k_f2b<<<(8192 + 255) / 256, 256, 0, stream>>>(F(35), WV,  8192);
    k_f2b<<<(8192 + 255) / 256, 256, 0, stream>>>(F(40), WO,  8192);
    k_rpeb<<<256, 256, 0, stream>>>(F(45), RPEB);

    // ---- bottleneck conv (feats -> comb[.,128..255]) ----
    k_conv3_480<<<512, 256, 0, stream>>>(F(0), WTB, SC + 0, COMB);

    // ---- decoder: dec1 (BN+ReLU) -> h ; dec2 (+bias) -> preds1 ----
    k_gemm_rm<4><<<512, 256, 0, stream>>>(COMB, 256, 128, WD1, 128, SC + 256, SC + 320, 0,
                                          HBUF, 64, 0, nullptr, nullptr);
    k_gemm_rm<1><<<512, 256, 0, stream>>>(HBUF, 64, 0, WD2, 64, nullptr, nullptr, 1,
                                          nullptr, 0, 0, P1, F(12));

    // ---- full-resolution spatial gather + recover -> ctxpix (value feats) ----
    k_sm_stats<<<24, 256, 0, stream>>>(P1, GMAX, GSUM);
    k_zero<<<1, 256, 0, stream>>>(CTXG, 3072);
    k_gather_global<<<64, 256, 0, stream>>>(P1, GMAX, GSUM, COMB, CTXG);
    k_recover_global<<<1024, 256, 0, stream>>>(P1, CTXG, CTXPIX);

    // ---- patch-level gather -> plc (key feats) ; channel gate -> qf ----
    k_patch_gather<<<256, 256, 0, stream>>>(P1, COMB, PLC);
    k_cattn<<<256, 256, 0, stream>>>(COMB, F(13), F(14), QF);

    // ---- q/k/v projections (WMMA, fused BN+ReLU) ----
    k_gemm_rm<4><<<512, 256, 0, stream>>>(QF, 128, 0, WQ1, 128, SC + 384, SC + 448, 0,
                                          T64, 64, 0, nullptr, nullptr);
    k_gemm_rm<4><<<512, 256, 0, stream>>>(T64, 64, 0, WQ2, 64, SC + 512, SC + 576, 0,
                                          QB, 64, 0, nullptr, nullptr);
    k_gemm_rm<4><<<512, 256, 0, stream>>>(PLC, 128, 0, WK1, 128, SC + 640, SC + 704, 0,
                                          T64, 64, 0, nullptr, nullptr);
    k_gemm_rm<4><<<512, 256, 0, stream>>>(T64, 64, 0, WK2, 64, SC + 768, SC + 832, 0,
                                          KB, 64, 0, nullptr, nullptr);
    k_gemm_rm<4><<<512, 256, 0, stream>>>(CTXPIX, 128, 0, WV, 128, SC + 896, SC + 960, 0,
                                          VB, 64, 0, nullptr, nullptr);

    // ---- RPE bias GEMM + attention + out projection ----
    k_pos_gemm<<<dim3(256, 2, 2), 256, 0, stream>>>(QB, RPEB, POSB);
    k_attn<<<512, 256, 278528, stream>>>(QB, KB, VB, POSB, CTXA);
    k_gemm_rm<8><<<512, 256, 0, stream>>>(CTXA, 64, 0, WO, 64, SC + 1024, SC + 1152, 0,
                                          COMB, 256, 0, nullptr, nullptr);  // res slice

    // ---- cat conv3x3 over [res|feats] -> preds2 (fp32 NCHW) ----
    k_conv3_cat<<<512, 256, 0, stream>>>(COMB, WTC, SC + 1280, OUT2);
}
